// CA_18090402251245
// MI455X (gfx1250) — matrix operand order repaired and checked by reading, not compiled, not run on hardware
//
#include <hip/hip_runtime.h>

typedef __attribute__((ext_vector_type(16))) _Float16 v16h;
typedef __attribute__((ext_vector_type(8)))  float    v8f;
typedef __attribute__((ext_vector_type(4)))  int      v4i;

typedef __attribute__((address_space(1))) v4i* gv4i_p;   // global
typedef __attribute__((address_space(3))) v4i* lv4i_p;   // LDS

#define Bn   16
#define Cn   8
#define Hn   512
#define Wn   512
#define HIDn 16

#define HAVE_ASYNC_LDS  __has_builtin(__builtin_amdgcn_global_load_async_to_lds_b128)
#define HAVE_WAIT_ASYNC __has_builtin(__builtin_amdgcn_s_wait_asynccnt)

static __device__ __forceinline__ unsigned pack2h(_Float16 a, _Float16 b) {
    union { _Float16 h[2]; unsigned u; } t;
    t.h[0] = a; t.h[1] = b;
    return t.u;
}

// One workgroup (256 threads = 8 waves, wave32) handles one (b, h) output row.
// LDS holds 3 input rows x 8 channels (wrap in H/W via &511).
// Per 16-pixel group:
//   WMMA1: Y1[16x16] = W1[16x32] x Perception[32x16]   (K=32 exact)
//   WMMA2: Y2[16x16] = W2pad[16x32] x Y1pad[32x16]     (K padded with zeros)
__global__ __launch_bounds__(256, 1) void CA_nca_step_kernel(
    const float* __restrict__ x,
    const float* __restrict__ w1w,   // [16, 32]
    const float* __restrict__ w1b,   // [16]
    const float* __restrict__ w2w,   // [8, 16]
    const int*   __restrict__ mask,  // [B,1,H,W]
    float*       __restrict__ out)   // [B,C,H,W]
{
    __shared__ float lds[Cn][3][Wn];   // 48 KB

    const int tid  = threadIdx.x;
    const int lane = tid & 31;
    const int wave = tid >> 5;
    const int hi   = lane >> 4;        // lane half: selects K range / D-row half
    const int ln   = lane & 15;        // column (pixel) / matrix row index

    const int bh = blockIdx.x;         // B*H blocks
    const int b  = bh >> 9;
    const int h  = bh & (Hn - 1);

    // ---- Stage x[b, :, h-1..h+1, :] into LDS (wrap in H) ----
    // 12288 floats = 3072 x 16B transfers; async Global->LDS (ASYNCcnt) when
    // available so no VGPR round-trip is needed.
    {
        const int total4 = Cn * 3 * (Wn / 4);      // 3072 x float4
        for (int i = tid; i < total4; i += 256) {
            const int w4 = i & (Wn / 4 - 1);       // 0..127
            const int cr = i >> 7;                 // 0..23
            const int c  = cr & 7;
            const int r  = cr >> 3;                // 0..2
            const int hh = (h + r - 1) & (Hn - 1);
            const size_t base = (((size_t)(b * Cn + c)) * Hn + hh) * Wn;
#if HAVE_ASYNC_LDS
            const float* g = x + base + (size_t)w4 * 4;
            void* l = (void*)&lds[c][r][w4 * 4];
            __builtin_amdgcn_global_load_async_to_lds_b128(
                (gv4i_p)(v4i*)(void*)const_cast<float*>(g),
                (lv4i_p)(v4i*)l,
                /*offset=*/0, /*cpol=*/0);
#else
            const float4 v = ((const float4*)(x + base))[w4];
            ((float4*)&lds[c][r][0])[w4] = v;
#endif
        }
#if HAVE_ASYNC_LDS
#if HAVE_WAIT_ASYNC
        __builtin_amdgcn_s_wait_asynccnt(0);
#else
        asm volatile("s_wait_asynccnt 0x0" ::: "memory");
#endif
#endif
    }

    // ---- WMMA A operands (weights) in contiguous K order (same as the B operands) ----
    // Lane holds row M = ln; half index i -> K = i + 16*hi.
    v16h a1, a2;
    {
        const int o = ln;
#pragma unroll
        for (int i = 0; i < 16; ++i) {
            const int k = i + 16 * hi;
            a1[i] = (_Float16)w1w[o * 32 + k];
            a2[i] = (_Float16)((o < 8 && k < 16) ? w2w[o * 16 + k] : 0.0f);
        }
    }
    // Bias for this lane's D rows: row m = r + 8*hi
    float bias[8];
#pragma unroll
    for (int r = 0; r < 8; ++r) bias[r] = w1b[r + 8 * hi];

    __syncthreads();

    // ---- 32 groups of 16 pixels; 8 waves -> 4 groups each ----
    for (int g = wave; g < Wn / 16; g += 8) {
        const int wp = g * 16 + ln;                 // this lane's pixel column
        const int wm = (wp - 1) & (Wn - 1);
        const int wq = (wp + 1) & (Wn - 1);

        // Perception -> B1 operand. Half i holds K = i + 16*hi; K = c*4 + f.
        v16h b1v;
#pragma unroll
        for (int cc = 0; cc < 4; ++cc) {
            const int c = 4 * hi + cc;
            const float s00 = lds[c][0][wm], s01 = lds[c][0][wp], s02 = lds[c][0][wq];
            const float s10 = lds[c][1][wm], s11 = lds[c][1][wp], s12 = lds[c][1][wq];
            const float s20 = lds[c][2][wm], s21 = lds[c][2][wp], s22 = lds[c][2][wq];
            const float ident = s11;
            const float sx  = (s02 - s00 + 2.0f * (s12 - s10) + s22 - s20) * 0.125f;
            const float sy  = (s20 - s00 + 2.0f * (s21 - s01) + s22 - s02) * 0.125f;
            const float lap = (s00 + s02 + s20 + s22
                               + 2.0f * (s01 + s10 + s12 + s21)
                               - 12.0f * s11) * 0.0625f;
            b1v[cc * 4 + 0] = (_Float16)ident;
            b1v[cc * 4 + 1] = (_Float16)sx;
            b1v[cc * 4 + 2] = (_Float16)sy;
            b1v[cc * 4 + 3] = (_Float16)lap;
        }

        // WMMA1: Y1 = W1 x P   (f32 accum)
        v8f acc = {};
        acc = __builtin_amdgcn_wmma_f32_16x16x32_f16(
            false, a1, false, b1v, (short)0, acc, false, false);

        // bias + relu + f16-convert, then PACK PAIRS so the half-wave exchange
        // costs 4 shuffles (dwords) instead of 8.
        unsigned own[4];
#pragma unroll
        for (int r = 0; r < 4; ++r) {
            float t0 = acc[2 * r + 0] + bias[2 * r + 0];
            float t1 = acc[2 * r + 1] + bias[2 * r + 1];
            t0 = t0 > 0.0f ? t0 : 0.0f;
            t1 = t1 > 0.0f ? t1 : 0.0f;
            own[r] = pack2h((_Float16)t0, (_Float16)t1);
        }

        // B2 operand: lanes 0-15 need Y1 rows 0..15 (rows 8..15 live in lane
        // L+16); lanes 16-31 hold K=16..31 which is the zero pad.
        union { unsigned u[8]; v16h v; } b2;
#pragma unroll
        for (int r = 0; r < 4; ++r) {
            const unsigned p = __shfl_xor(own[r], 16, 32);
            b2.u[r]     = hi ? 0u : own[r];   // K rows 0..7
            b2.u[r + 4] = hi ? 0u : p;        // K rows 8..15
        }

        // WMMA2: Y2 = W2pad x Y1pad
        v8f acc2 = {};
        acc2 = __builtin_amdgcn_wmma_f32_16x16x32_f16(
            false, a2, false, b2.v, (short)0, acc2, false, false);

        // Lanes 0-15 hold D rows 0..7 = the 8 output channels of pixel wp.
        if (!hi) {
            const float mf = (float)mask[((size_t)b * Hn + h) * Wn + wp];
#pragma unroll
            for (int ch = 0; ch < 8; ++ch) {
                const size_t idx = (((size_t)(b * Cn + ch)) * Hn + h) * Wn + wp;
                out[idx] = lds[ch][1][wp] + acc2[ch] * mf;
            }
        }
    }
}

extern "C" void kernel_launch(void* const* d_in, const int* in_sizes, int n_in,
                              void* d_out, int out_size, void* d_ws, size_t ws_size,
                              hipStream_t stream) {
    const float* x    = (const float*)d_in[0];
    const float* w1w  = (const float*)d_in[1];
    const float* w1b  = (const float*)d_in[2];
    const float* w2w  = (const float*)d_in[3];
    const int*   msk  = (const int*)d_in[4];
    float* out = (float*)d_out;

    dim3 grid(Bn * Hn);   // 8192 workgroups, one output row each
    dim3 block(256);      // 8 waves (wave32)
    CA_nca_step_kernel<<<grid, block, 0, stream>>>(x, w1w, w1b, w2w, msk, out);
}
